// EulerProductMoE_25005299597547
// MI455X (gfx1250) — compile-verified, matrix operand order checked by reading, not run on hardware
//
#include <hip/hip_runtime.h>
#include <hip/hip_bf16.h>

typedef __bf16 bf16;
typedef __attribute__((ext_vector_type(16))) __bf16 v16bf;
typedef __attribute__((ext_vector_type(8)))  float  v8f;
typedef __attribute__((ext_vector_type(4)))  unsigned int v4u;
typedef __attribute__((ext_vector_type(4)))  int v4i;

#define IN_DIM  1024
#define HID_DIM 512
#define OUT_DIM 1024
#define NE      6
#define BTOK    32768
#define BM      64      // tokens per workgroup
#define KC      32      // K-chunk (one WMMA K step)
#define BLK     512     // 16 waves per workgroup

// ---------------- ws layout (bytes) ----------------
#define WS_XBF   ((size_t)0)                        // 32768*1024*2 = 64 MiB
#define WS_W1T   ((size_t)67108864)                 // 6*512*1024*2 = 6 MiB, [e][n][k]
#define WS_W2T   ((size_t)73400320)                 // 6*1024*512*2 = 6 MiB, [e][o][h]
#define WS_GATES ((size_t)79691776)                 // 32768*6*4
// total ~76.8 MB

// ---------------- async copy helpers (CDNA5 ASYNCcnt path) ----------------
#if defined(__has_builtin)
#if __has_builtin(__builtin_amdgcn_global_load_async_to_lds_b128)
#define HAVE_ASYNC_LDS 1
#endif
#endif

#ifdef HAVE_ASYNC_LDS
typedef __attribute__((address_space(1))) v4i* gptr_v4i;
typedef __attribute__((address_space(3))) v4i* lptr_v4i;
#endif

__device__ __forceinline__ void async_cp16(void* lds_dst, const void* g_src) {
#ifdef HAVE_ASYNC_LDS
    __builtin_amdgcn_global_load_async_to_lds_b128(
        (gptr_v4i)(v4i*)(void*)(g_src),
        (lptr_v4i)(v4i*)(lds_dst), 0, 0);
#else
    *(v4u*)lds_dst = *(const v4u*)g_src;
#endif
}

// wait until at most N of this wave's async ops remain outstanding
// (in-order completion => all older ops are done)
template<int N>
__device__ __forceinline__ void wait_async() {
#ifdef HAVE_ASYNC_LDS
#if __has_builtin(__builtin_amdgcn_s_wait_asynccnt)
    __builtin_amdgcn_s_wait_asynccnt(N);
#else
    asm volatile("s_wait_asynccnt %0" :: "i"(N) : "memory");
#endif
#endif
}

// ---------------- prep kernels ----------------
__global__ void cvt_x_kernel(const float* __restrict__ x, bf16* __restrict__ xb, int n) {
    int i = blockIdx.x * blockDim.x + threadIdx.x;
    int stride = gridDim.x * blockDim.x;
    for (; i < n; i += stride) xb[i] = (bf16)x[i];
}

// w1: [e][k][n] (k stride HID_DIM) -> w1T: [e][n][k] (n stride IN_DIM)
__global__ void cvt_w1_kernel(const float* __restrict__ w1, bf16* __restrict__ w1t) {
    int n = NE * IN_DIM * HID_DIM;
    int i = blockIdx.x * blockDim.x + threadIdx.x;
    int stride = gridDim.x * blockDim.x;
    for (; i < n; i += stride) {
        int e = i / (IN_DIM * HID_DIM);
        int r = i % (IN_DIM * HID_DIM);
        int k = r / HID_DIM;
        int c = r % HID_DIM;
        w1t[(size_t)e * IN_DIM * HID_DIM + (size_t)c * IN_DIM + k] = (bf16)w1[i];
    }
}

// w2: [e][h][o] (h stride OUT_DIM) -> w2T: [e][o][h] (o stride HID_DIM)
__global__ void cvt_w2_kernel(const float* __restrict__ w2, bf16* __restrict__ w2t) {
    int n = NE * HID_DIM * OUT_DIM;
    int i = blockIdx.x * blockDim.x + threadIdx.x;
    int stride = gridDim.x * blockDim.x;
    for (; i < n; i += stride) {
        int e = i / (HID_DIM * OUT_DIM);
        int r = i % (HID_DIM * OUT_DIM);
        int h = r / OUT_DIM;
        int o = r % OUT_DIM;
        w2t[(size_t)e * HID_DIM * OUT_DIM + (size_t)o * HID_DIM + h] = (bf16)w2[i];
    }
}

__global__ void gate_kernel(const float* __restrict__ x,
                            const float* __restrict__ g2w, const float* __restrict__ g2b,
                            const float* __restrict__ g3w, const float* __restrict__ g3b,
                            float* __restrict__ gates) {
    int t = blockIdx.x * blockDim.x + threadIdx.x;
    if (t >= BTOK) return;
    const float* xr = x + (size_t)t * IN_DIM;
    float a2 = 0.f, s0 = 0.f, s1 = 0.f, s2 = 0.f;
    for (int i = 0; i < IN_DIM; ++i) {
        float xv = xr[i];
        a2 += xv * g2w[i];
        s0 += xv * g3w[i * 3 + 0];
        s1 += xv * g3w[i * 3 + 1];
        s2 += xv * g3w[i * 3 + 2];
    }
    a2 += g2b[0]; s0 += g3b[0]; s1 += g3b[1]; s2 += g3b[2];
    float alpha = 1.f / (1.f + __expf(-a2));
    float m = fmaxf(s0, fmaxf(s1, s2));
    float e0 = __expf(s0 - m), e1 = __expf(s1 - m), e2 = __expf(s2 - m);
    float inv = 1.f / (e0 + e1 + e2);
    float p0 = e0 * inv, p1 = e1 * inv, p2 = e2 * inv;
    float ga = 1.f - alpha;
    float* gr = gates + (size_t)t * NE;
    gr[0] = ga * p0;    gr[1] = ga * p1;    gr[2] = ga * p2;
    gr[3] = alpha * p0; gr[4] = alpha * p1; gr[5] = alpha * p2;
}

// ---------------- fused MoE kernel ----------------
// LDS layout (dynamic):
//   HS   : [BM][HID_DIM] bf16  64 KB @ 0
//   GW   : [BM][NE] f32  1.5 KB     @ 65536
//   BUF0 : W[<=1024][KC] 64 KB + X[BM][KC] 4 KB  @ 67584
//   BUF1 : same                                   @ 137216
#define SM_GW      65536
#define SM_B0      67584
#define SM_BSTRIDE 69632    // byte distance between BUF0 and BUF1
#define SM_BX      65536    // byte offset of X region inside a buffer
#define SM_TOTAL   206848

// async ops issued per wave by each stage flavor (must match stage bodies)
#define C_STAGE1 5          // 4x W1 b128 + 1x X b128
#define C_STAGE2 8          // 8x W2 b128

__global__ __launch_bounds__(BLK)
void moe_fused_kernel(const bf16* __restrict__ xbf,
                      const bf16* __restrict__ w1T, const float* __restrict__ b1,
                      const bf16* __restrict__ w2T, const float* __restrict__ b2,
                      const float* __restrict__ gates,
                      float* __restrict__ out) {
    extern __shared__ char smem[];
    bf16*  HS = (bf16*)smem;
    float* GW = (float*)(smem + SM_GW);

    const int tid  = threadIdx.x;
    const int lane = tid & 31;
    const int wv   = tid >> 5;       // 0..15
    const int lr   = lane & 15;      // row/col within 16
    const int kh   = (lane >> 4) * 16;
    const int lh   = lane >> 4;      // 0/1 half-select for C rows
    const int r0   = blockIdx.x * BM;

    const bf16* xrow = xbf + (size_t)r0 * IN_DIM;

    // buffer pointers computed arithmetically (no pointer arrays -> no
    // addrspacecast static initializers)
    auto bufW = [&](int b) -> bf16* {
        return (bf16*)(smem + SM_B0 + (unsigned)b * SM_BSTRIDE);
    };
    auto bufX = [&](int b) -> bf16* {
        return (bf16*)(smem + SM_B0 + SM_BX + (unsigned)b * SM_BSTRIDE);
    };

    // per-thread staging coordinates (constant across rounds)
    const int sn0  = tid >> 2;        // 0..127 : W row group
    const int sseg = (tid & 3) * 8;   // bf16 offset within 32-elem row
    const int xq   = tid & 255;       // X chunk id (duplicated across half-blocks)
    const int xr   = xq >> 2;         // X row 0..63
    const int xs   = (xq & 3) * 8;    // X col offset

    // stage1: W1 chunk [512][KC] + X chunk [BM][KC] into buffer b
    auto stage1 = [&](int b, int k0, const bf16* w1e) {
        bf16* bw = bufW(b);
        bf16* bx = bufX(b);
        #pragma unroll
        for (int j = 0; j < 4; ++j) {
            int n = sn0 + j * 128;
            async_cp16(bw + n * KC + sseg, w1e + (size_t)n * IN_DIM + k0 + sseg);
        }
        async_cp16(bx + xr * KC + xs, xrow + (size_t)xr * IN_DIM + k0 + xs);
    };
    // stage2: W2 chunk [1024][KC] into buffer b
    auto stage2 = [&](int b, int k0, const bf16* w2e) {
        bf16* bw = bufW(b);
        #pragma unroll
        for (int j = 0; j < 8; ++j) {
            int o = sn0 + j * 128;
            async_cp16(bw + o * KC + sseg, w2e + (size_t)o * HID_DIM + k0 + sseg);
        }
    };

    if (tid < BM * NE) GW[tid] = gates[(size_t)r0 * NE + tid];

    // persistent output accumulators: 4 m-tiles x 4 n-tiles per wave
    v8f acc2[4][4];
    const v8f vzero = {0.f,0.f,0.f,0.f,0.f,0.f,0.f,0.f};
    #pragma unroll
    for (int mt = 0; mt < 4; ++mt)
        #pragma unroll
        for (int nt = 0; nt < 4; ++nt) acc2[mt][nt] = vzero;

    // prologue: prefetch expert-0 GEMM1 round-0 into buf0
    stage1(0, 0, w1T);
    __syncthreads();   // GW visible; async still in flight

    for (int e = 0; e < NE; ++e) {
        const bf16* w1e = w1T + (size_t)e * IN_DIM * HID_DIM;   // [n][k]
        const bf16* w2e = w2T + (size_t)e * HID_DIM * OUT_DIM;  // [o][h]

        // ---------------- GEMM1: H = relu(X @ W1 + b1) * gate ----------------
        v8f acc1[4][2];
        #pragma unroll
        for (int mt = 0; mt < 4; ++mt)
            #pragma unroll
            for (int nt = 0; nt < 2; ++nt) acc1[mt][nt] = vzero;

        for (int kk = 0; kk < IN_DIM / KC; ++kk) {
            const int cur = kk & 1, nxt = cur ^ 1;
            // prefetch next chunk into the other buffer, then drain the current one
            if (kk < IN_DIM / KC - 1) {
                stage1(nxt, (kk + 1) * KC, w1e);
                wait_async<C_STAGE1>();
            } else {
                stage2(nxt, 0, w2e);          // cross-phase prefetch: GEMM2 round 0
                wait_async<C_STAGE2>();
            }
            __syncthreads();                  // current buffer visible to all

            const bf16* bw = bufW(cur);
            const bf16* bx = bufX(cur);
            v16bf a[4], bfr[2];
            #pragma unroll
            for (int mt = 0; mt < 4; ++mt)
                a[mt] = *(const v16bf*)(bx + (mt * 16 + lr) * KC + kh);
            const int nb = wv * 32;
            #pragma unroll
            for (int nt = 0; nt < 2; ++nt)
                bfr[nt] = *(const v16bf*)(bw + (nb + nt * 16 + lr) * KC + kh);
            #pragma unroll
            for (int mt = 0; mt < 4; ++mt)
                #pragma unroll
                for (int nt = 0; nt < 2; ++nt)
                    acc1[mt][nt] = __builtin_amdgcn_wmma_f32_16x16x32_bf16(
                        false, a[mt], false, bfr[nt], (short)0, acc1[mt][nt], false, false);
            __syncthreads();                  // all reads done before buffer is re-DMA'd
        }

        // epilogue: bias + relu + gate-scale -> HS (bf16); overlaps GEMM2 prefetch
        #pragma unroll
        for (int mt = 0; mt < 4; ++mt) {
            #pragma unroll
            for (int nt = 0; nt < 2; ++nt) {
                const int ncol = wv * 32 + nt * 16 + lr;
                const float b1v = b1[e * HID_DIM + ncol];
                #pragma unroll
                for (int i = 0; i < 8; ++i) {
                    const int row = mt * 16 + i + lh * 8;
                    float v = acc1[mt][nt][i] + b1v;
                    v = v > 0.f ? v : 0.f;
                    v *= GW[row * NE + e];
                    HS[row * HID_DIM + ncol] = (bf16)v;
                }
            }
        }
        __syncthreads();

        // ---------------- GEMM2: out += H' @ W2 (chained C across experts) ----
        for (int kk = 0; kk < HID_DIM / KC; ++kk) {
            const int cur = kk & 1, nxt = cur ^ 1;
            if (kk < HID_DIM / KC - 1) {
                stage2(nxt, (kk + 1) * KC, w2e);
                wait_async<C_STAGE2>();
            } else if (e < NE - 1) {
                stage1(nxt, 0, w1e + IN_DIM * HID_DIM);   // next expert's first chunk
                wait_async<C_STAGE1>();
            } else {
                wait_async<0>();
            }
            __syncthreads();

            const bf16* bw = bufW(cur);
            v16bf a[4], bfr[4];
            #pragma unroll
            for (int mt = 0; mt < 4; ++mt)
                a[mt] = *(const v16bf*)(HS + (mt * 16 + lr) * HID_DIM + kk * KC + kh);
            const int ob = wv * 64;
            #pragma unroll
            for (int nt = 0; nt < 4; ++nt)
                bfr[nt] = *(const v16bf*)(bw + (ob + nt * 16 + lr) * KC + kh);
            #pragma unroll
            for (int mt = 0; mt < 4; ++mt)
                #pragma unroll
                for (int nt = 0; nt < 4; ++nt)
                    acc2[mt][nt] = __builtin_amdgcn_wmma_f32_16x16x32_bf16(
                        false, a[mt], false, bfr[nt], (short)0, acc2[mt][nt], false, false);
            __syncthreads();
        }
    }

    // ---------------- final epilogue: add sum_e gw*b2[e], store f32 ----------
    #pragma unroll
    for (int mt = 0; mt < 4; ++mt) {
        #pragma unroll
        for (int nt = 0; nt < 4; ++nt) {
            const int o = wv * 64 + nt * 16 + lr;
            float b2v[NE];
            #pragma unroll
            for (int e = 0; e < NE; ++e) b2v[e] = b2[e * OUT_DIM + o];
            #pragma unroll
            for (int i = 0; i < 8; ++i) {
                const int row = mt * 16 + i + lh * 8;
                float v = acc2[mt][nt][i];
                #pragma unroll
                for (int e = 0; e < NE; ++e) v += GW[row * NE + e] * b2v[e];
                out[(size_t)(r0 + row) * OUT_DIM + o] = v;
            }
        }
    }
}

// ---------------- host ----------------
extern "C" void kernel_launch(void* const* d_in, const int* in_sizes, int n_in,
                              void* d_out, int out_size, void* d_ws, size_t ws_size,
                              hipStream_t stream) {
    const float* x    = (const float*)d_in[0];
    const float* g2_w = (const float*)d_in[1];
    const float* g2_b = (const float*)d_in[2];
    const float* g3_w = (const float*)d_in[3];
    const float* g3_b = (const float*)d_in[4];
    const float* w1   = (const float*)d_in[5];
    const float* b1   = (const float*)d_in[6];
    const float* w2   = (const float*)d_in[7];
    const float* b2   = (const float*)d_in[8];
    float* out = (float*)d_out;

    char* ws = (char*)d_ws;
    bf16*  xbf   = (bf16*)(ws + WS_XBF);
    bf16*  w1t   = (bf16*)(ws + WS_W1T);
    bf16*  w2t   = (bf16*)(ws + WS_W2T);
    float* gates = (float*)(ws + WS_GATES);

    cvt_x_kernel<<<4096, 256, 0, stream>>>(x, xbf, BTOK * IN_DIM);
    cvt_w1_kernel<<<4096, 256, 0, stream>>>(w1, w1t);
    cvt_w2_kernel<<<4096, 256, 0, stream>>>(w2, w2t);
    gate_kernel<<<BTOK / 256, 256, 0, stream>>>(x, g2_w, g2_b, g3_w, g3_b, gates);

    (void)hipFuncSetAttribute((const void*)moe_fused_kernel,
                              hipFuncAttributeMaxDynamicSharedMemorySize, SM_TOTAL);

    moe_fused_kernel<<<BTOK / BM, BLK, SM_TOTAL, stream>>>(
        xbf, w1t, b1, w2t, b2, gates, out);
}
